// EnsembleObsHead_3642132267038
// MI455X (gfx1250) — matrix operand order repaired and checked by reading, not compiled, not run on hardware
//
#include <hip/hip_runtime.h>

// ---------------------------------------------------------------------------
// EnsembleObsHead on MI455X (gfx1250): bf16 WMMA path, fp32 accumulate,
// async global->LDS staging (ASYNCcnt), NT stores for the streaming output.
// M=8, E=512, H=1024, V=4096, BI=4096.
// ---------------------------------------------------------------------------

typedef __bf16 bf16;
typedef __attribute__((ext_vector_type(16))) __bf16 v16bf;
typedef __attribute__((ext_vector_type(8)))  __bf16 v8bf;
typedef __attribute__((ext_vector_type(4)))  __bf16 v4bf;
typedef __attribute__((ext_vector_type(8)))  float  v8f;
typedef __attribute__((ext_vector_type(4)))  float  v4f;
typedef __attribute__((ext_vector_type(4)))  int    v4i;

#define LDSK 40   // padded LDS row stride (elems) for 32-wide K tiles: 80B = 20 dwords
                  // -> conflict-free b128 fragment reads across wave32

#if __has_builtin(__builtin_amdgcn_global_load_async_to_lds_b128)
#define HAS_ASYNC 1
#endif

#if __has_builtin(__builtin_amdgcn_s_wait_asynccnt)
#define ASYNC_WAIT(n) __builtin_amdgcn_s_wait_asynccnt(n)
#else
#define ASYNC_WAIT(n) asm volatile("s_wait_asynccnt %0" ::"i"(n) : "memory")
#endif

#ifdef HAS_ASYNC
typedef __attribute__((address_space(1))) v4i* gptr_v4i;  // global (prints as __device__)
typedef __attribute__((address_space(3))) v4i* lptr_v4i;  // LDS
// 16B global -> LDS async copy (per-lane addresses), tracked by ASYNCcnt.
static __device__ __forceinline__ void async_ld16(const bf16* g, bf16* l) {
    __builtin_amdgcn_global_load_async_to_lds_b128(
        (gptr_v4i)const_cast<bf16*>(g), (lptr_v4i)l, 0, 0);
}
#endif

static __device__ __forceinline__ v8f wmma_bf16(v16bf a, v16bf b, v8f c) {
    return __builtin_amdgcn_wmma_f32_16x16x32_bf16(
        /*neg_a=*/false, a, /*neg_b=*/false, b,
        /*c_mod=*/(short)0, c, /*reuse_a=*/false, /*reuse_b=*/false);
}

// A fragment (16x32 bf16), per ISA layout:
// lanes 0-15: row M=lane, K 0..7 (V0-3) and 16..23 (V4-7)
// lanes16-31: row M=lane-16, K 8..15 and 24..31
static __device__ __forceinline__ v16bf load_frag_a(const bf16* base, int lane) {
    const int row  = lane & 15;
    const int half = lane >> 4;
    const bf16* p = base + row * LDSK + half * 8;
    union { v16bf v; v8bf h[2]; } u;
    u.h[0] = *(const v8bf*)(p);
    u.h[1] = *(const v8bf*)(p + 16);
    return u.v;
}

// B fragment (32x16 bf16) read from a transposed [N][K] LDS tile:
// lanes 0-15: col N=lane, K 0..7 and 16..23 ; lanes 16-31: col N=lane-16, K 8..15 and 24..31
static __device__ __forceinline__ v16bf load_frag_b(const bf16* base, int lane) {
    const int col  = lane & 15;
    const int half = lane >> 4;
    const bf16* p = base + col * LDSK + half * 8;
    union { v16bf v; v8bf h[2]; } u;
    u.h[0] = *(const v8bf*)(p);
    u.h[1] = *(const v8bf*)(p + 16);
    return u.v;
}

// ---------------------------------------------------------------------------
// Kernel 0: fp32 [R][C] -> bf16 transposed [C][R], per member (blockIdx.z).
// Sources are read exactly once -> non-temporal loads.
// ---------------------------------------------------------------------------
__global__ __launch_bounds__(256) void cvt_transpose(
    const float* __restrict__ src, bf16* __restrict__ dst, int R, int C)
{
    __shared__ float tile[32][33];
    const int t = threadIdx.x;
    const size_t moff = (size_t)blockIdx.z * (size_t)R * (size_t)C;
    const int rb = blockIdx.y * 32, cb = blockIdx.x * 32;

    {
        const int r = t >> 3, c4 = (t & 7) * 4;
        const v4f v = __builtin_nontemporal_load(
            (const v4f*)(src + moff + (size_t)(rb + r) * C + cb + c4));
        tile[r][c4 + 0] = v[0]; tile[r][c4 + 1] = v[1];
        tile[r][c4 + 2] = v[2]; tile[r][c4 + 3] = v[3];
    }
    __syncthreads();
    {
        const int c = t >> 3, r4 = (t & 7) * 4;
        v4bf o;
        o[0] = (bf16)tile[r4 + 0][c];
        o[1] = (bf16)tile[r4 + 1][c];
        o[2] = (bf16)tile[r4 + 2][c];
        o[3] = (bf16)tile[r4 + 3][c];
        *(v4bf*)(dst + moff + (size_t)(cb + c) * R + rb + r4) = o;
    }
}

// ---------------------------------------------------------------------------
// Kernel 1: h = SiLU(LayerNorm(x @ W1 + b1)) -> bf16 [M*BI][H]
// Block: 32 rows x full H=1024.  8 waves; wave w owns col tiles
// {ct*128 + w*16 : ct=0..7} so each staged 256-col chunk feeds all waves.
// ---------------------------------------------------------------------------
__global__ __launch_bounds__(256) void gemm1_ln_silu(
    const float* __restrict__ x,     // [M*BI][E=512] fp32
    const bf16*  __restrict__ W1t,   // [M][H=1024][E=512] bf16 (transposed)
    const float* __restrict__ b1,    // [M][H]
    const float* __restrict__ lnw,   // [M][H]
    const float* __restrict__ lnb,   // [M][H]
    bf16* __restrict__ hout)         // [M*BI][H] bf16
{
    __shared__ bf16  lA[32 * LDSK];      // 32 rows  x 32 K
    __shared__ bf16  lB[256 * LDSK];     // 256 cols x 32 K (transposed chunk)
    __shared__ float psum[8][32];
    __shared__ float psq[8][32];
    __shared__ float smean[32];
    __shared__ float srstd[32];

    const int t    = threadIdx.x;
    const int lane = t & 31;
    const int w    = t >> 5;
    const int m    = blockIdx.y;
    const int row0 = blockIdx.x * 32;

    const float* xb = x + ((size_t)(m * 4096 + row0)) * 512;
    const bf16*  wb = W1t + (size_t)m * 1024 * 512;

    v8f acc[2][8];
    #pragma unroll
    for (int rt = 0; rt < 2; ++rt)
        #pragma unroll
        for (int ct = 0; ct < 8; ++ct) acc[rt][ct] = (v8f)0.0f;

    const int ar = t >> 3, ac4 = (t & 7) * 4;   // A staging: 4 floats/thread

    for (int it = 0; it < 16; ++it) {
        const int k0 = it * 32;
        for (int c = 0; c < 4; ++c) {
            __syncthreads();
            if (c == 0) {  // stage A: 32x32 fp32 -> bf16 (register path: needs cvt)
                const v4f v = __builtin_nontemporal_load(
                    (const v4f*)(xb + (size_t)ar * 512 + k0 + ac4));
                v4bf o; o[0] = (bf16)v[0]; o[1] = (bf16)v[1]; o[2] = (bf16)v[2]; o[3] = (bf16)v[3];
                *(v4bf*)(lA + ar * LDSK + ac4) = o;
            }
            {   // stage Bt chunk: rows [c*256, c*256+256) of W1t, 64B per thread
                const bf16* g = wb + (size_t)(c * 256 + t) * 512 + k0;
                bf16* d = lB + t * LDSK;
#ifdef HAS_ASYNC
                async_ld16(g + 0,  d + 0);
                async_ld16(g + 8,  d + 8);
                async_ld16(g + 16, d + 16);
                async_ld16(g + 24, d + 24);
#else
                *(v8bf*)(d + 0)  = *(const v8bf*)(g + 0);
                *(v8bf*)(d + 8)  = *(const v8bf*)(g + 8);
                *(v8bf*)(d + 16) = *(const v8bf*)(g + 16);
                *(v8bf*)(d + 24) = *(const v8bf*)(g + 24);
#endif
            }
#ifdef HAS_ASYNC
            ASYNC_WAIT(0);
#endif
            __syncthreads();
            const v16bf a0 = load_frag_a(lA + 0 * 16 * LDSK, lane);
            const v16bf a1 = load_frag_a(lA + 1 * 16 * LDSK, lane);
            #pragma unroll
            for (int j = 0; j < 2; ++j) {
                const int ct = 2 * c + j;
                const v16bf b = load_frag_b(lB + (j * 128 + w * 16) * LDSK, lane);
                acc[0][ct] = wmma_bf16(a0, b, acc[0][ct]);
                acc[1][ct] = wmma_bf16(a1, b, acc[1][ct]);
            }
        }
    }

    // ---- epilogue: +b1, LayerNorm over H=1024, SiLU, bf16 store ----
    const int lc = lane & 15, half = lane >> 4;

    v8f s[2], q[2];
    s[0] = (v8f)0.0f; s[1] = (v8f)0.0f; q[0] = (v8f)0.0f; q[1] = (v8f)0.0f;
    #pragma unroll
    for (int ct = 0; ct < 8; ++ct) {
        const int col = ct * 128 + w * 16 + lc;
        const float bv = b1[m * 1024 + col];
        #pragma unroll
        for (int rt = 0; rt < 2; ++rt) {
            v8f a = acc[rt][ct];
            #pragma unroll
            for (int g = 0; g < 8; ++g) a[g] += bv;
            acc[rt][ct] = a;
            s[rt] += a;
            q[rt] += a * a;
        }
    }
    // reduce over the 16 lanes of each half-wave (C layout: halves hold rows g / g+8)
    #pragma unroll
    for (int mask = 1; mask < 16; mask <<= 1) {
        #pragma unroll
        for (int rt = 0; rt < 2; ++rt)
            #pragma unroll
            for (int g = 0; g < 8; ++g) {
                s[rt][g] += __shfl_xor(s[rt][g], mask, 32);
                q[rt][g] += __shfl_xor(q[rt][g], mask, 32);
            }
    }
    if (lc == 0) {  // lanes 0 and 16 publish per-row partials for this wave
        #pragma unroll
        for (int rt = 0; rt < 2; ++rt)
            #pragma unroll
            for (int g = 0; g < 8; ++g) {
                const int r = rt * 16 + half * 8 + g;
                psum[w][r] = s[rt][g];
                psq[w][r]  = q[rt][g];
            }
    }
    __syncthreads();
    if (t < 32) {
        float ss = 0.0f, qq = 0.0f;
        #pragma unroll
        for (int ww = 0; ww < 8; ++ww) { ss += psum[ww][t]; qq += psq[ww][t]; }
        const float mean = ss * (1.0f / 1024.0f);
        const float var  = qq * (1.0f / 1024.0f) - mean * mean;
        smean[t] = mean;
        srstd[t] = rsqrtf(var + 1e-5f);
    }
    __syncthreads();

    float mv[2][8], rv[2][8];
    #pragma unroll
    for (int rt = 0; rt < 2; ++rt)
        #pragma unroll
        for (int g = 0; g < 8; ++g) {
            const int r = rt * 16 + half * 8 + g;
            mv[rt][g] = smean[r];
            rv[rt][g] = srstd[r];
        }
    #pragma unroll
    for (int ct = 0; ct < 8; ++ct) {
        const int col = ct * 128 + w * 16 + lc;
        const float gw = lnw[m * 1024 + col];
        const float gb = lnb[m * 1024 + col];
        #pragma unroll
        for (int rt = 0; rt < 2; ++rt) {
            const int rbase = m * 4096 + row0 + rt * 16 + half * 8;
            #pragma unroll
            for (int g = 0; g < 8; ++g) {
                float v = (acc[rt][ct][g] - mv[rt][g]) * rv[rt][g] * gw + gb;
                v = v * (1.0f / (1.0f + __expf(-v)));   // SiLU
                hout[(size_t)(rbase + g) * 1024 + col] = (bf16)v;
            }
        }
    }
}

// ---------------------------------------------------------------------------
// Kernel 2: out = h @ W2 + b2   (per member: 4096x4096x1024)
// Block tile 128x128, BK=32, 8 waves as 2x4, wave tile 64x32 ->
// 8 v_wmma per wave per K-step.
// Async path: 3-deep LDS pipeline, GLOBAL_LOAD_ASYNC_TO_LDS_B128 staging,
// s_wait_asynccnt retires stage `it` while stage `it+1` stays in flight.
// ---------------------------------------------------------------------------
__global__ __launch_bounds__(256) void gemm2(
    const bf16* __restrict__ hbuf,   // [M*BI][H] bf16
    const bf16* __restrict__ W2t,    // [M][V][H] bf16 (transposed)
    const float* __restrict__ b2,    // [M][V]
    float* __restrict__ out)         // [M*BI][V]
{
    const int t    = threadIdx.x;
    const int lane = t & 31;
    const int w    = t >> 5;
    const int wm   = w >> 2;     // 0..1 (64 rows each)
    const int wn   = w & 3;      // 0..3 (32 cols each)
    const int m    = blockIdx.z;
    const int row0 = blockIdx.y * 128;
    const int n0   = blockIdx.x * 128;

    const bf16* Ab = hbuf + ((size_t)(m * 4096 + row0)) * 1024;
    const bf16* Bb = W2t + (size_t)m * 4096 * 1024 + (size_t)n0 * 1024;

    const int srow = t >> 1;     // 0..127
    const int sseg = t & 1;      // 32B half of the 64B row slab

    v8f acc[4][2];
    #pragma unroll
    for (int rt = 0; rt < 4; ++rt) { acc[rt][0] = (v8f)0.0f; acc[rt][1] = (v8f)0.0f; }

#ifdef HAS_ASYNC
    __shared__ bf16 lA[3][128 * LDSK];
    __shared__ bf16 lB[3][128 * LDSK];

    auto stage = [&](int buf, int it) {
        const int k0 = it * 32;
        const bf16* pa = Ab + (size_t)srow * 1024 + k0 + sseg * 16;
        bf16* da = &lA[buf][srow * LDSK + sseg * 16];
        async_ld16(pa, da);
        async_ld16(pa + 8, da + 8);
        const bf16* pb = Bb + (size_t)srow * 1024 + k0 + sseg * 16;
        bf16* db = &lB[buf][srow * LDSK + sseg * 16];
        async_ld16(pb, db);
        async_ld16(pb + 8, db + 8);
    };

    stage(0, 0);
    stage(1, 1);
    for (int it = 0; it < 32; ++it) {
        // retire stage `it` (4 async ops per stage per thread stay in flight for it+1)
        if (it + 1 < 32) { ASYNC_WAIT(4); } else { ASYNC_WAIT(0); }
        __syncthreads();           // all lanes' stage-`it` LDS writes visible;
                                   // all stage-(it-1) fragment reads retired
        if (it + 2 < 32) stage((it + 2) % 3, it + 2);

        const int buf = it % 3;
        v16bf bf0 = load_frag_b(&lB[buf][(wn * 32 + 0)  * LDSK], lane);
        v16bf bf1 = load_frag_b(&lB[buf][(wn * 32 + 16) * LDSK], lane);
        #pragma unroll
        for (int rt = 0; rt < 4; ++rt) {
            const v16bf a = load_frag_a(&lA[buf][(wm * 64 + rt * 16) * LDSK], lane);
            acc[rt][0] = wmma_bf16(a, bf0, acc[rt][0]);
            acc[rt][1] = wmma_bf16(a, bf1, acc[rt][1]);
        }
    }
#else
    __shared__ bf16 lA[2][128 * LDSK];
    __shared__ bf16 lB[2][128 * LDSK];
    v8bf rA0, rA1, rB0, rB1;
    {
        const bf16* pa = Ab + (size_t)srow * 1024 + sseg * 16;
        rA0 = *(const v8bf*)(pa);     rA1 = *(const v8bf*)(pa + 8);
        const bf16* pb = Bb + (size_t)srow * 1024 + sseg * 16;
        rB0 = *(const v8bf*)(pb);     rB1 = *(const v8bf*)(pb + 8);
        bf16* da = &lA[0][srow * LDSK + sseg * 16];
        *(v8bf*)(da) = rA0; *(v8bf*)(da + 8) = rA1;
        bf16* db = &lB[0][srow * LDSK + sseg * 16];
        *(v8bf*)(db) = rB0; *(v8bf*)(db + 8) = rB1;
    }
    for (int it = 0; it < 32; ++it) {
        __syncthreads();
        if (it + 1 < 32) {
            const int k0 = (it + 1) * 32;
            const bf16* pa = Ab + (size_t)srow * 1024 + k0 + sseg * 16;
            rA0 = *(const v8bf*)(pa);     rA1 = *(const v8bf*)(pa + 8);
            const bf16* pb = Bb + (size_t)srow * 1024 + k0 + sseg * 16;
            rB0 = *(const v8bf*)(pb);     rB1 = *(const v8bf*)(pb + 8);
        }
        const int buf = it & 1;
        v16bf bf0 = load_frag_b(&lB[buf][(wn * 32 + 0)  * LDSK], lane);
        v16bf bf1 = load_frag_b(&lB[buf][(wn * 32 + 16) * LDSK], lane);
        #pragma unroll
        for (int rt = 0; rt < 4; ++rt) {
            const v16bf a = load_frag_a(&lA[buf][(wm * 64 + rt * 16) * LDSK], lane);
            acc[rt][0] = wmma_bf16(a, bf0, acc[rt][0]);
            acc[rt][1] = wmma_bf16(a, bf1, acc[rt][1]);
        }
        if (it + 1 < 32) {
            const int nb = buf ^ 1;
            bf16* da = &lA[nb][srow * LDSK + sseg * 16];
            *(v8bf*)(da) = rA0; *(v8bf*)(da + 8) = rA1;
            bf16* db = &lB[nb][srow * LDSK + sseg * 16];
            *(v8bf*)(db) = rB0; *(v8bf*)(db + 8) = rB1;
        }
    }
#endif

    // epilogue: + b2, fp32 non-temporal store (write-once 512MB stream;
    // keep W2t/h resident in L2 instead)
    const int lc = lane & 15, half = lane >> 4;
    #pragma unroll
    for (int ct = 0; ct < 2; ++ct) {
        const int col = n0 + wn * 32 + ct * 16 + lc;
        const float bv = b2[m * 4096 + col];
        #pragma unroll
        for (int rt = 0; rt < 4; ++rt) {
            const int rbase = m * 4096 + row0 + wm * 64 + rt * 16 + half * 8;
            #pragma unroll
            for (int g = 0; g < 8; ++g)
                __builtin_nontemporal_store(acc[rt][ct][g] + bv,
                                            &out[(size_t)(rbase + g) * 4096 + col]);
        }
    }
}

// ---------------------------------------------------------------------------
extern "C" void kernel_launch(void* const* d_in, const int* in_sizes, int n_in,
                              void* d_out, int out_size, void* d_ws, size_t ws_size,
                              hipStream_t stream) {
    (void)in_sizes; (void)n_in; (void)out_size; (void)ws_size;

    const float* x   = (const float*)d_in[0];   // [32768][512]
    const float* W1  = (const float*)d_in[1];   // [8][512][1024]
    const float* b1  = (const float*)d_in[2];   // [8][1024]
    const float* lnw = (const float*)d_in[3];   // [8][1024]
    const float* lnb = (const float*)d_in[4];   // [8][1024]
    const float* W2  = (const float*)d_in[5];   // [8][1024][4096]
    const float* b2  = (const float*)d_in[6];   // [8][4096]
    float* out = (float*)d_out;                 // [32768][4096]

    char* ws = (char*)d_ws;
    bf16* hbuf = (bf16*)(ws);                                   // 64 MB
    bf16* W2t  = (bf16*)(ws + (size_t)64 * 1024 * 1024);        // 64 MB
    bf16* W1t  = (bf16*)(ws + (size_t)128 * 1024 * 1024);       //  8 MB

    // W1 [E=512][H=1024] -> W1t [H][E] bf16 ; W2 [H=1024][V=4096] -> W2t [V][H] bf16
    cvt_transpose<<<dim3(1024 / 32, 512 / 32, 8), 256, 0, stream>>>(W1, W1t, 512, 1024);
    cvt_transpose<<<dim3(4096 / 32, 1024 / 32, 8), 256, 0, stream>>>(W2, W2t, 1024, 4096);

    gemm1_ln_silu<<<dim3(4096 / 32, 8), 256, 0, stream>>>(x, W1t, b1, lnw, lnb, hbuf);
    gemm2<<<dim3(4096 / 128, 4096 / 128, 8), 256, 0, stream>>>(hbuf, W2t, b2, out);
}
